// Sequence_53300544143404
// MI455X (gfx1250) — matrix operand order repaired and checked by reading, not compiled, not run on hardware
//
#include <hip/hip_runtime.h>

typedef __attribute__((ext_vector_type(16))) _Float16 v16h;
typedef __attribute__((ext_vector_type(8)))  _Float16 v8h;
typedef __attribute__((ext_vector_type(8)))  float    v8f;

#define HDIM 51
#define G4   204      // 4*H
#define G4P  208      // padded gate count (13*16)
#define KP   64       // K padded for 2 x k32 WMMA tiles
#define GST  208      // gate-interleaved row stride (51*4 -> pad 208)
#define NT   13       // N tiles of 16 covering 204
#define TLEN 1000
#define BATCH 4096

__device__ __forceinline__ float sigf(float x) { return 1.0f / (1.0f + __expf(-x)); }

__device__ __forceinline__ v8f wmma_f16(v16h a, v16h b, v8f c) {
    // D = A(16x32,f16) * B(32x16,f16) + C(16x16,f32)
    return __builtin_amdgcn_wmma_f32_16x16x32_f16(false, a, false, b, (short)0, c, false, false);
}

// A-fragment (16x32 f16): lane m=l&15, hi=l>>4; element e -> K = 16*hi + e
__device__ __forceinline__ v16h load_afrag(const _Float16* hrow, int kt, int hi) {
    union { v16h v; v8h h[2]; } u;
    u.h[0] = *(const v8h*)(hrow + kt * 32 + hi * 16);
    u.h[1] = *(const v8h*)(hrow + kt * 32 + hi * 16 + 8);
    return u.v;
}

// B-fragment (32x16 f16): lane n=l&15, lanes 16-31 hold K 16-31; 16 contiguous f16 per lane
__device__ __forceinline__ v16h load_bfrag(const _Float16* w, int nt, int kt, int lane) {
    int n  = nt * 16 + (lane & 15);
    int k0 = kt * 32 + (lane >> 4) * 16;
    return *(const v16h*)(w + n * KP + k0);
}

__global__ __launch_bounds__(32, 1) void lstm2_seq_wmma(
    const float* __restrict__ inp,
    const float* __restrict__ Wih1, const float* __restrict__ Whh1,
    const float* __restrict__ bih1, const float* __restrict__ bhh1,
    const float* __restrict__ Wih2, const float* __restrict__ Whh2,
    const float* __restrict__ bih2, const float* __restrict__ bhh2,
    const float* __restrict__ Wlin, const float* __restrict__ blinP,
    const int* __restrict__ futP,
    float* __restrict__ out)
{
    __shared__ __align__(32) _Float16 sW1[G4P * KP];   // Whh1 (f16, K- and N-padded)
    __shared__ __align__(32) _Float16 sW2[G4P * KP];   // Wih2
    __shared__ __align__(32) _Float16 sW3[G4P * KP];   // Whh2
    __shared__ __align__(16) float sB1v[G4];           // biases, gate-interleaved [jj*4+q]
    __shared__ __align__(16) float sXWv[G4];           // x-input weights, gate-interleaved
    __shared__ __align__(16) float sB2v[G4];
    __shared__ float sWL[52];
    __shared__ __align__(16) float sGate[16 * GST];    // gate tile, interleaved [m][jj*4+q]
    __shared__ __align__(32) _Float16 sH1[16 * KP];
    __shared__ __align__(32) _Float16 sH2[16 * KP];
    __shared__ float sXB[16];                          // autoregressive feedback

    const int tid = threadIdx.x;   // 0..31, single wave per workgroup

    // ---- one-time weight staging (fp32 -> f16, K/N padded with zeros) ----
    for (int idx = tid; idx < G4P * KP; idx += 32) {
        int n = idx >> 6, k = idx & 63;
        bool ok = (n < G4) && (k < HDIM);
        sW1[idx] = (_Float16)(ok ? Whh1[n * HDIM + k] : 0.f);
        sW2[idx] = (_Float16)(ok ? Wih2[n * HDIM + k] : 0.f);
        sW3[idx] = (_Float16)(ok ? Whh2[n * HDIM + k] : 0.f);
    }
    for (int n = tid; n < G4; n += 32) {
        int q = n / HDIM, jj = n - q * HDIM;    // gate, column
        sB1v[jj * 4 + q] = bih1[n] + bhh1[n];
        sB2v[jj * 4 + q] = bih2[n] + bhh2[n];
        sXWv[jj * 4 + q] = Wih1[n];             // W_ih1 is [204][1]
    }
    for (int j = tid; j < 52; j += 32) sWL[j] = (j < HDIM) ? Wlin[j] : 0.f;
    for (int idx = tid; idx < 16 * KP; idx += 32) {
        sH1[idx] = (_Float16)0.f;
        sH2[idx] = (_Float16)0.f;
    }
    if (tid < 16) sXB[tid] = 0.f;
    __syncthreads();   // S_NOP for single-wave workgroup; DScnt orders staging vs loop

    const int lane = tid;
    const int m    = lane & 15;
    const int hi   = lane >> 4;
    const int row0 = blockIdx.x * 16;
    const float blin = blinP[0];
    const int fut  = futP[0];
    const int S    = TLEN + fut;

    float c1r[26], c2r[26];
    #pragma unroll
    for (int u = 0; u < 26; ++u) { c1r[u] = 0.f; c2r[u] = 0.f; }

    v16h a1[2], a2[2];
    #pragma unroll
    for (int kt = 0; kt < 2; ++kt) {
        #pragma unroll
        for (int i = 0; i < 16; ++i) { a1[kt][i] = (_Float16)0.f; a2[kt][i] = (_Float16)0.f; }
    }

    const float* gm = &sGate[m * GST];
    _Float16* h1row = &sH1[m * KP];
    _Float16* h2row = &sH2[m * KP];
    const int col   = lane & 15;

    for (int s = 0; s < S; ++s) {
        // ======== layer 1: gates = h1 @ Whh1^T  (WMMA) ========
        for (int nt = 0; nt < NT; ++nt) {
            v8f acc = {};
            acc = wmma_f16(a1[0], load_bfrag(sW1, nt, 0, lane), acc);
            acc = wmma_f16(a1[1], load_bfrag(sW1, nt, 1, lane), acc);
            int n = nt * 16 + col;
            if (n < G4) {                       // drop padding columns
                int q = n / HDIM, jj = n - q * HDIM;
                int c0 = jj * 4 + q;            // gate-interleaved slot
                #pragma unroll
                for (int r = 0; r < 8; ++r) sGate[(r + 8 * hi) * GST + c0] = acc[r];
            }
        }
        // ======== layer 1 elementwise: +bias +x*Wih1, cell update ========
        float x = (s < TLEN) ? inp[(row0 + m) * TLEN + s] : sXB[m];
        for (int u = 0; u < 26; ++u) {
            int j = hi * 26 + u;
            if (j < HDIM) {
                float4 g  = *(const float4*)(gm   + 4 * j);
                float4 b  = *(const float4*)(sB1v + 4 * j);
                float4 wx = *(const float4*)(sXWv + 4 * j);
                float gi = g.x + b.x + x * wx.x;
                float gf = g.y + b.y + x * wx.y;
                float gg = g.z + b.z + x * wx.z;
                float go = g.w + b.w + x * wx.w;
                float cn = sigf(gf) * c1r[u] + sigf(gi) * tanhf(gg);
                c1r[u] = cn;
                h1row[j] = (_Float16)(sigf(go) * tanhf(cn));
            }
        }
        a1[0] = load_afrag(h1row, 0, hi);
        a1[1] = load_afrag(h1row, 1, hi);

        // ======== layer 2: gates = h1_new @ Wih2^T + h2 @ Whh2^T ========
        for (int nt = 0; nt < NT; ++nt) {
            v8f acc = {};
            acc = wmma_f16(a1[0], load_bfrag(sW2, nt, 0, lane), acc);
            acc = wmma_f16(a1[1], load_bfrag(sW2, nt, 1, lane), acc);
            acc = wmma_f16(a2[0], load_bfrag(sW3, nt, 0, lane), acc);
            acc = wmma_f16(a2[1], load_bfrag(sW3, nt, 1, lane), acc);
            int n = nt * 16 + col;
            if (n < G4) {
                int q = n / HDIM, jj = n - q * HDIM;
                int c0 = jj * 4 + q;
                #pragma unroll
                for (int r = 0; r < 8; ++r) sGate[(r + 8 * hi) * GST + c0] = acc[r];
            }
        }
        // ======== layer 2 elementwise + output projection ========
        float part = 0.f;
        for (int u = 0; u < 26; ++u) {
            int j = hi * 26 + u;
            if (j < HDIM) {
                float4 g = *(const float4*)(gm   + 4 * j);
                float4 b = *(const float4*)(sB2v + 4 * j);
                float gi = g.x + b.x;
                float gf = g.y + b.y;
                float gg = g.z + b.z;
                float go = g.w + b.w;
                float cn = sigf(gf) * c2r[u] + sigf(gi) * tanhf(gg);
                c2r[u] = cn;
                float hn = sigf(go) * tanhf(cn);
                h2row[j] = (_Float16)hn;
                part += hn * sWL[j];
            }
        }
        a2[0] = load_afrag(h2row, 0, hi);
        a2[1] = load_afrag(h2row, 1, hi);

        part += __shfl_xor(part, 16, 32);     // combine the two j-halves per row
        float ov = part + blin;
        if (hi == 0) {
            out[(size_t)(row0 + m) * (size_t)S + s] = ov;
            sXB[m] = ov;                      // feedback for autoregressive phase
        }
    }
}

extern "C" void kernel_launch(void* const* d_in, const int* in_sizes, int n_in,
                              void* d_out, int out_size, void* d_ws, size_t ws_size,
                              hipStream_t stream) {
    (void)in_sizes; (void)n_in; (void)d_ws; (void)ws_size; (void)out_size;
    lstm2_seq_wmma<<<dim3(BATCH / 16), dim3(32), 0, stream>>>(
        (const float*)d_in[0],
        (const float*)d_in[1], (const float*)d_in[2],
        (const float*)d_in[3], (const float*)d_in[4],
        (const float*)d_in[5], (const float*)d_in[6],
        (const float*)d_in[7], (const float*)d_in[8],
        (const float*)d_in[9], (const float*)d_in[10],
        (const int*)d_in[11],
        (float*)d_out);
}
